// FullAttention_45732811768064
// MI455X (gfx1250) — matrix operand order repaired and checked by reading, not compiled, not run on hardware
//
#include <hip/hip_runtime.h>

typedef __attribute__((ext_vector_type(16))) _Float16 v16h;
typedef __attribute__((ext_vector_type(8)))  _Float16 v8h;
typedef __attribute__((ext_vector_type(8)))  float    v8f;

#define B_ 4
#define T_ 2048
#define S_ 2048
#define H_ 16
#define E_ 64
#define NEG_INF (-1e30f)
#define NCH (S_ / 64)

__device__ inline v16h load_frag16(const _Float16* p) {
    v16h r;
#pragma unroll
    for (int i = 0; i < 16; ++i) r[i] = p[i];
    return r;
}

// Flash-attention forward, no-max streaming softmax in log2 domain.
//  - Q pre-scaled by log2(e)/sqrt(E): one v_exp_f32 per score element.
//  - Row sums l = P x ones on the matrix pipe (extra WMMA), not VALU.
//  - K/V double-buffered in LDS; one barrier per 64-key chunk.
//  - key_padding_mask is a prefix mask (arange < length): break out of the S-loop at the
//    first fully-masked chunk (~25% work saved on average), and only the single boundary
//    chunk pays the per-element mask bias.
// One block = one (b,h) x 128 query rows (8 waves x 16 rows).
__global__ __launch_bounds__(256)
void attn_kernel(const float* __restrict__ Q, const float* __restrict__ K,
                 const float* __restrict__ V, const unsigned char* __restrict__ mask,
                 _Float16* __restrict__ AO)
{
    const int bh   = blockIdx.x;
    const int b    = bh >> 4;       // H_=16
    const int h    = bh & 15;
    const int qt   = blockIdx.y;    // 0..15

    const int tid  = threadIdx.x;
    const int wave = tid >> 5;
    const int lane = tid & 31;
    const int hl   = lane & 15;
    const int hi   = lane >> 4;

    // Padded strides (72 halves = 36 words, gcd(36,64)=4 -> conflict-free row access).
    __shared__ _Float16 Ksh[2][64][72];    // [buf][key][dim]
    __shared__ _Float16 Vt [2][64][72];    // [buf][dim][key] (transposed)
    __shared__ _Float16 Psh[8][16][72];    // per-wave P staging [qrow][key]

    const int qrow0 = qt * 128 + wave * 16;

    // ---- Q A-fragments (16x32 f16), scaled by log2(e)/sqrt(E)
    v16h aQ0, aQ1;
    {
        const float qs = 0.125f * 1.44269504089f;
        const float* qbase = Q + ((size_t)(b * T_) + qrow0 + hl) * (H_ * E_) + h * E_;
#pragma unroll
        for (int idx = 0; idx < 16; ++idx) {
            const int kk = hi * 16 + idx;
            aQ0[idx] = (_Float16)(qs * qbase[kk]);
            aQ1[idx] = (_Float16)(qs * qbase[32 + kk]);
        }
    }

    // Constant B fragment: column 0 all-ones -> D[:,0] = row sums of A.
    v16h bOnes;
    {
        const _Float16 o1 = (hl == 0) ? (_Float16)1.0f : (_Float16)0.0f;
#pragma unroll
        for (int i = 0; i < 16; ++i) bOnes[i] = o1;
    }

    v8f oacc[4];
#pragma unroll
    for (int n = 0; n < 4; ++n) oacc[n] = (v8f){};
    v8f lacc = (v8f){};

    const size_t kvStride = (size_t)(H_ * E_);
    const float* Kbh = K + (size_t)b * S_ * kvStride + h * E_;
    const float* Vbh = V + (size_t)b * S_ * kvStride + h * E_;
    const unsigned char* mb = mask + (size_t)b * S_;

    // Staging coordinates: each thread handles one key row x 16 dims, K and V.
    const int skey = tid >> 2;          // 0..63
    const int sd0  = (tid & 3) * 16;    // 0,16,32,48

    float4 rk[4], rv[4];
    auto loadchunk = [&](int s0) {
        const float4* gk = (const float4*)(Kbh + (size_t)(s0 + skey) * kvStride + sd0);
        const float4* gv = (const float4*)(Vbh + (size_t)(s0 + skey) * kvStride + sd0);
#pragma unroll
        for (int j = 0; j < 4; ++j) { rk[j] = gk[j]; rv[j] = gv[j]; }
    };
    auto storechunk = [&](int buf) {
        v8h p0, p1;
        p0[0]=(_Float16)rk[0].x; p0[1]=(_Float16)rk[0].y; p0[2]=(_Float16)rk[0].z; p0[3]=(_Float16)rk[0].w;
        p0[4]=(_Float16)rk[1].x; p0[5]=(_Float16)rk[1].y; p0[6]=(_Float16)rk[1].z; p0[7]=(_Float16)rk[1].w;
        p1[0]=(_Float16)rk[2].x; p1[1]=(_Float16)rk[2].y; p1[2]=(_Float16)rk[2].z; p1[3]=(_Float16)rk[2].w;
        p1[4]=(_Float16)rk[3].x; p1[5]=(_Float16)rk[3].y; p1[6]=(_Float16)rk[3].z; p1[7]=(_Float16)rk[3].w;
        *(v8h*)&Ksh[buf][skey][sd0]     = p0;
        *(v8h*)&Ksh[buf][skey][sd0 + 8] = p1;
        _Float16 (*vt)[72] = Vt[buf];
        vt[sd0 + 0][skey] = (_Float16)rv[0].x;  vt[sd0 + 1][skey] = (_Float16)rv[0].y;
        vt[sd0 + 2][skey] = (_Float16)rv[0].z;  vt[sd0 + 3][skey] = (_Float16)rv[0].w;
        vt[sd0 + 4][skey] = (_Float16)rv[1].x;  vt[sd0 + 5][skey] = (_Float16)rv[1].y;
        vt[sd0 + 6][skey] = (_Float16)rv[1].z;  vt[sd0 + 7][skey] = (_Float16)rv[1].w;
        vt[sd0 + 8][skey] = (_Float16)rv[2].x;  vt[sd0 + 9][skey] = (_Float16)rv[2].y;
        vt[sd0 +10][skey] = (_Float16)rv[2].z;  vt[sd0 +11][skey] = (_Float16)rv[2].w;
        vt[sd0 +12][skey] = (_Float16)rv[3].x;  vt[sd0 +13][skey] = (_Float16)rv[3].y;
        vt[sd0 +14][skey] = (_Float16)rv[3].z;  vt[sd0 +15][skey] = (_Float16)rv[3].w;
    };

    loadchunk(0);
    storechunk(0);
    __syncthreads();

    for (int i = 0; i < NCH; ++i) {
        const int s0 = i * 64;
        // Prefix mask: first fully-masked chunk => all remaining keys masked. Uniform branch.
        if (mb[s0] == 0) break;

        const int cur = i & 1;
        if (i + 1 < NCH) loadchunk(s0 + 64);   // global loads in flight during compute

        const _Float16 (*Kc)[72] = Ksh[cur];
        const _Float16 (*Vc)[72] = Vt[cur];
        _Float16 (*Pw)[72] = Psh[wave];

        // ---- Scores: four 16x16 tiles over 64 keys (2 WMMA each, K-dim 64 = 2x32)
        v8f sc[4];
#pragma unroll
        for (int t = 0; t < 4; ++t) {
            v16h b0 = load_frag16(&Kc[t * 16 + hl][hi * 16]);
            v16h b1 = load_frag16(&Kc[t * 16 + hl][32 + hi * 16]);
            v8f s = (v8f){};
            s = __builtin_amdgcn_wmma_f32_16x16x32_f16(false, aQ0, false, b0, (short)0, s, false, false);
            s = __builtin_amdgcn_wmma_f32_16x16x32_f16(false, aQ1, false, b1, (short)0, s, false, false);
            sc[t] = s;
        }

        // ---- exp2; only the single boundary chunk needs the per-element mask bias
        if (mb[s0 + 63] != 0) {
#pragma unroll
            for (int t = 0; t < 4; ++t)
#pragma unroll
                for (int r = 0; r < 8; ++r)
                    Pw[hi * 8 + r][t * 16 + hl] = (_Float16)__builtin_amdgcn_exp2f(sc[t][r]);
        } else {
#pragma unroll
            for (int t = 0; t < 4; ++t) {
                const float bias = (mb[s0 + t * 16 + hl] != 0) ? 0.f : NEG_INF;
#pragma unroll
                for (int r = 0; r < 8; ++r)
                    Pw[hi * 8 + r][t * 16 + hl] = (_Float16)__builtin_amdgcn_exp2f(sc[t][r] + bias);
            }
        }

        // ---- O += P(16x64) x V(64x64); l += P x ones (matrix pipe, column 0)
#pragma unroll
        for (int kq = 0; kq < 2; ++kq) {
            v16h aP;
#pragma unroll
            for (int idx = 0; idx < 16; ++idx) {
                const int kk = hi * 16 + idx;
                aP[idx] = Pw[hl][kq * 32 + kk];
            }
            lacc = __builtin_amdgcn_wmma_f32_16x16x32_f16(false, aP, false, bOnes, (short)0, lacc, false, false);
#pragma unroll
            for (int n = 0; n < 4; ++n) {
                v16h bV = load_frag16(&Vc[n * 16 + hl][kq * 32 + hi * 16]);
                oacc[n] = __builtin_amdgcn_wmma_f32_16x16x32_f16(false, aP, false, bV, (short)0, oacc[n], false, false);
            }
        }

        if (i + 1 < NCH) storechunk(1 - cur);
        __syncthreads();
    }

    // ---- Row sums live in column 0 (lane hi*16); broadcast, normalize, store
    float linv[8];
#pragma unroll
    for (int r = 0; r < 8; ++r) linv[r] = 1.0f / __shfl(lacc[r], hi * 16, 32);

#pragma unroll
    for (int n = 0; n < 4; ++n) {
#pragma unroll
        for (int r = 0; r < 8; ++r) {
            const int q = qrow0 + hi * 8 + r;
            AO[((size_t)(b * T_) + q) * (H_ * E_) + h * E_ + n * 16 + hl] =
                (_Float16)(oacc[n][r] * linv[r]);
        }
    }
}

// Output projection: out(8192,64) = X(8192,1024) @ W(1024,64) + bias. Deterministic tiled GEMM.
__global__ __launch_bounds__(256)
void proj_kernel(const _Float16* __restrict__ X, const float* __restrict__ W,
                 const float* __restrict__ bias, float* __restrict__ out)
{
    __shared__ float Wsh[128][64];
    const int row = blockIdx.x * 64 + (threadIdx.x >> 2);
    const int cg  = (threadIdx.x & 3) * 16;
    float acc[16];
#pragma unroll
    for (int j = 0; j < 16; ++j) acc[j] = 0.f;

    for (int k0 = 0; k0 < H_ * E_; k0 += 128) {
        __syncthreads();
        for (int i = threadIdx.x; i < 128 * 64; i += 256)
            Wsh[i >> 6][i & 63] = W[(size_t)(k0 + (i >> 6)) * E_ + (i & 63)];
        __syncthreads();
        const _Float16* xrow = X + (size_t)row * (H_ * E_) + k0;
        for (int k = 0; k < 128; ++k) {
            const float xv = (float)xrow[k];
#pragma unroll
            for (int j = 0; j < 16; ++j) acc[j] += xv * Wsh[k][cg + j];
        }
    }
#pragma unroll
    for (int j = 0; j < 16; ++j)
        out[(size_t)row * E_ + cg + j] = acc[j] + bias[cg + j];
}

extern "C" void kernel_launch(void* const* d_in, const int* in_sizes, int n_in,
                              void* d_out, int out_size, void* d_ws, size_t ws_size,
                              hipStream_t stream) {
    const float* Q  = (const float*)d_in[0];
    const float* K  = (const float*)d_in[1];
    const float* V  = (const float*)d_in[2];
    const unsigned char* mask = (const unsigned char*)d_in[3]; // bool (B,S), nonzero = keep
    const float* Wout = (const float*)d_in[4];
    const float* bout = (const float*)d_in[5];

    _Float16* attn = (_Float16*)d_ws; // B*T*H*E f16 = 16 MiB scratch

    attn_kernel<<<dim3(B_ * H_, T_ / 128), 256, 0, stream>>>(Q, K, V, mask, attn);
    proj_kernel<<<dim3((B_ * T_) / 64), 256, 0, stream>>>(attn, Wout, bout, (float*)d_out);
}
